// Asym_Attention_39092792328572
// MI455X (gfx1250) — matrix operand order repaired and checked by reading, not compiled, not run on hardware
//
#include <hip/hip_runtime.h>
#include <hip/hip_bf16.h>

// CDNA5 / gfx1250: wave32; WMMA 16x16x32 bf16 + f16 (f32 accum).
typedef __attribute__((ext_vector_type(16))) __bf16    v16bf;
typedef __attribute__((ext_vector_type(16))) _Float16  v16h;
typedef __attribute__((ext_vector_type(8)))  float     v8f;

#define B1   64
#define B2T  128          // 2*B (concat of V and I streams)
#define NTOK 384
#define CDIM 768
#define NH   12
#define DH   64
#define LMT  128          // t_h*t_w*2
#define MTOT (B2T*NTOK)   // 49152 rows in the big GEMMs
#define QKVC (3*CDIM)     // 2304

// GEMM macro-tile: 64 (M) x 128 (N), 8 waves, each wave 16x64 (4 WMMAs/k-step)
#define GM   64
#define GN   128
#define G_STR 40          // LDS row stride (bf16) for A/B tiles: 80B, 16B aligned

// Attention LDS strides (elements) -> conflict-free fragment reads.
#define KT_STR 72         // K chunk tile row stride (bf16), 144B
#define VT_STR 264        // V^T tile row stride (f16), 528B
#define SH_STR 520        // score row stride (f16), 1040B
#define KCH    256        // keys per staged chunk

// ---- WMMA fragment index helpers (contiguous k order used here: k = 16*hi + s) ----
__device__ __forceinline__ int a_kidx(int i, int hi) {   // A (MxK) lane layout
  return (hi << 4) + (i << 1);
}
__device__ __forceinline__ int b_kidx(int i, int hi) {   // B (KxN) lane layout
  return (hi << 4) + (i << 1);
}

__device__ __forceinline__ v8f wmma_bf16(v16bf a, v16bf b, v8f c) {
  return __builtin_amdgcn_wmma_f32_16x16x32_bf16(false, a, false, b, (short)0, c,
                                                 false, false);
}
__device__ __forceinline__ v8f wmma_f16(v16h a, v16h b, v8f c) {
  return __builtin_amdgcn_wmma_f32_16x16x32_f16(false, a, false, b, (short)0, c,
                                                false, false);
}

// Element offset of a K/V row for the asymmetric key layout.
__device__ __forceinline__ size_t key_off(int b2, int h, int idx, bool mt) {
  int sb2, tok;
  if (mt) { sb2 = b2; tok = idx; }
  else {
    int b = b2 & 63;
    if      (idx < LMT)     { sb2 = b;      tok = idx; }
    else if (idx < 2 * LMT) { sb2 = 64 + b; tok = idx - LMT; }
    else                    { sb2 = b2;     tok = idx - LMT; }
  }
  return (((size_t)sb2 * NH + h) * NTOK + tok) * (size_t)DH;
}

// ---------------- Kernel 1: QKV projection (64x128 macro-tile GEMM) ---------
// qkv[m][c] = sum_k x[m][k] w_qkv[c][k] + b_qkv[c]; scatter to Q/K (bf16) and
// V (f16) in [b2][h][tok][d] layout.
__global__ __launch_bounds__(256) void qkv_kernel(
    const float* __restrict__ x_v, const float* __restrict__ x_i,
    const float* __restrict__ w_qkv, const float* __restrict__ b_qkv,
    __bf16* __restrict__ Qb, __bf16* __restrict__ Kb, _Float16* __restrict__ Vh) {
  __shared__ __bf16 As[GM * G_STR];     // 5 KB  : A tile [m][k]
  __shared__ __bf16 Bs[GN * G_STR];     // 10 KB : B tile [n][k]
  const int tid  = threadIdx.x;
  const int lane = tid & 31, wid = tid >> 5;
  const int wm = wid >> 1, wn = wid & 1;           // 4x2 wave grid
  const int m0 = blockIdx.y * GM;
  const int c0 = blockIdx.x * GN;
  const int mrow = lane & 15, hi = lane >> 4, ncol = lane & 15;
  v8f acc0 = {}, acc1 = {}, acc2 = {}, acc3 = {};
  for (int k0 = 0; k0 < CDIM; k0 += 32) {
    for (int e = tid; e < GM * 32; e += 256) {     // stage A (x -> bf16)
      int r = e >> 5, c = e & 31;
      int m = m0 + r;
      float v = (m < B1 * NTOK) ? x_v[(size_t)m * CDIM + k0 + c]
                                : x_i[(size_t)(m - B1 * NTOK) * CDIM + k0 + c];
      As[r * G_STR + c] = (__bf16)v;
    }
    for (int e = tid; e < GN * 32; e += 256) {     // stage B (w -> bf16)
      int r = e >> 5, c = e & 31;
      Bs[r * G_STR + c] = (__bf16)w_qkv[(size_t)(c0 + r) * CDIM + k0 + c];
    }
    __syncthreads();
    v16bf a;
    const __bf16* apr = &As[(wm * 16 + mrow) * G_STR];
    for (int i = 0; i < 8; ++i) {
      int k = a_kidx(i, hi);
      a[2 * i]     = apr[k];
      a[2 * i + 1] = apr[k + 1];
    }
#pragma unroll
    for (int nt = 0; nt < 4; ++nt) {
      const __bf16* bp = &Bs[(wn * 64 + nt * 16 + ncol) * G_STR + hi * 16];
      v16bf b;
      for (int i = 0; i < 16; ++i) b[i] = bp[i];   // 16 contiguous = 2x b128
      v8f& acc = (nt == 0) ? acc0 : (nt == 1) ? acc1 : (nt == 2) ? acc2 : acc3;
      acc = wmma_bf16(a, b, acc);
    }
    __syncthreads();
  }
#pragma unroll
  for (int nt = 0; nt < 4; ++nt) {
    const v8f acc = (nt == 0) ? acc0 : (nt == 1) ? acc1 : (nt == 2) ? acc2 : acc3;
    const int col   = c0 + wn * 64 + nt * 16 + ncol;
    const float bias = b_qkv[col];
    const int which = col / CDIM, rem = col % CDIM;
    const int h = rem / DH, d = rem % DH;
    for (int j = 0; j < 8; ++j) {
      int m = m0 + wm * 16 + j + hi * 8;
      int b2 = m / NTOK, tok = m % NTOK;
      size_t idx = (((size_t)b2 * NH + h) * NTOK + tok) * DH + d;
      float r = acc[j] + bias;
      if      (which == 0) Qb[idx] = (__bf16)r;
      else if (which == 1) Kb[idx] = (__bf16)r;
      else                 Vh[idx] = (_Float16)r;
    }
  }
}

// ---------------- Kernel 2: asymmetric attention ---------------------------
// Block = 4 waves = 64 q rows of one (b2, head); K/V staged through LDS in
// 256-key chunks shared by all 4 waves.
__global__ __launch_bounds__(128) void attn_kernel(
    const __bf16* __restrict__ Qb, const __bf16* __restrict__ Kb,
    const _Float16* __restrict__ Vh, __bf16* __restrict__ ATT) {
  __shared__ __align__(16) unsigned char kvbuf[KCH * KT_STR * 2]; // 36 KB union
  __shared__ _Float16 Sh[64 * SH_STR];                            // 65 KB scores
  __bf16*   Kt = (__bf16*)kvbuf;      // [key][d]   stride KT_STR
  _Float16* Vt = (_Float16*)kvbuf;    // [d][key]   stride VT_STR (transposed)

  const int tid = threadIdx.x;
  const int lane = tid & 31, wid = tid >> 5;
  const int b2 = blockIdx.z, h = blockIdx.y;
  const int q0 = blockIdx.x * 64;
  const bool mt = (q0 < LMT);
  const int Lk = mt ? LMT : 4 * LMT;            // 128 or 512
  const int mrow = lane & 15, hi = lane >> 4, ncol = lane & 15;
  const float scale = 0.125f;                   // 1/sqrt(64)

  // Per-wave Q fragments (16x64 = two 16x32 A tiles).
  const __bf16* Qp = Qb + (((size_t)b2 * NH + h) * NTOK + q0 + wid * 16) * DH;
  v16bf aq0, aq1;
  for (int i = 0; i < 8; ++i) {
    int k = a_kidx(i, hi);
    aq0[2 * i]     = Qp[mrow * DH + k];
    aq0[2 * i + 1] = Qp[mrow * DH + k + 1];
    aq1[2 * i]     = Qp[mrow * DH + 32 + k];
    aq1[2 * i + 1] = Qp[mrow * DH + 32 + k + 1];
  }

  // ---- Phase 1: S = scale * Q K^T, chunked over keys --------------------
  for (int ch = 0; ch < Lk; ch += KCH) {
    const int cn = (Lk - ch < KCH) ? (Lk - ch) : KCH;
    for (int c = tid; c < cn * 8; c += 128) {     // stage K chunk (16B copies)
      int row = c >> 3, seg = c & 7;
      size_t off = key_off(b2, h, ch + row, mt);
      *(uint4*)&Kt[row * KT_STR + seg * 8] = *(const uint4*)(Kb + off + seg * 8);
    }
    __syncthreads();
    for (int kt = 0; kt < cn; kt += 16) {
      const __bf16* kr = &Kt[(kt + ncol) * KT_STR];
      v16bf b0, b1;
      for (int i = 0; i < 8; ++i) {
        int k = b_kidx(i, hi);
        b0[2 * i]     = kr[k];
        b0[2 * i + 1] = kr[k + 1];
        b1[2 * i]     = kr[32 + k];
        b1[2 * i + 1] = kr[32 + k + 1];
      }
      v8f c = {};
      c = wmma_bf16(aq0, b0, c);
      c = wmma_bf16(aq1, b1, c);
      for (int j = 0; j < 8; ++j)
        Sh[(wid * 16 + j + hi * 8) * SH_STR + ch + kt + ncol] =
            (_Float16)(c[j] * scale);
    }
    __syncthreads();
  }

  // ---- Phase 2: softmax in place (2 threads per row) --------------------
  {
    int r = tid >> 1, half = tid & 1;
    int nb = half * (Lk >> 1), ne = nb + (Lk >> 1);
    _Float16* sr = &Sh[r * SH_STR];
    float mx = -3.4e38f;
    for (int n = nb; n < ne; ++n) mx = fmaxf(mx, (float)sr[n]);
    mx = fmaxf(mx, __shfl_xor(mx, 1, 32));
    float sum = 0.f;
    for (int n = nb; n < ne; ++n) {
      float e = __expf((float)sr[n] - mx);
      sr[n] = (_Float16)e;
      sum += e;
    }
    sum += __shfl_xor(sum, 1, 32);
    float inv = 1.f / sum;
    for (int n = nb; n < ne; ++n) sr[n] = (_Float16)((float)sr[n] * inv);
  }
  __syncthreads();

  // ---- Phase 3: O = P V (f16 WMMA), V staged transposed in chunks -------
  v8f o0 = {}, o1 = {}, o2 = {}, o3 = {};
  for (int ch = 0; ch < Lk; ch += KCH) {
    const int cn = (Lk - ch < KCH) ? (Lk - ch) : KCH;
    for (int c = tid; c < cn * 8; c += 128) {
      int row = c >> 3, seg = c & 7;
      size_t off = key_off(b2, h, ch + row, mt);
      const _Float16* vr = Vh + off + seg * 8;
#pragma unroll
      for (int t = 0; t < 8; ++t) Vt[(seg * 8 + t) * VT_STR + row] = vr[t];
    }
    __syncthreads();
    for (int kt = 0; kt < cn; kt += 32) {
      v16h ap;
      const _Float16* pr = &Sh[(wid * 16 + mrow) * SH_STR + ch + kt];
      for (int i = 0; i < 8; ++i) {
        int k = a_kidx(i, hi);
        ap[2 * i]     = pr[k];
        ap[2 * i + 1] = pr[k + 1];
      }
#pragma unroll
      for (int nt = 0; nt < 4; ++nt) {
        const _Float16* vrow = &Vt[(nt * 16 + ncol) * VT_STR + kt + hi * 16];
        v16h bv;
        for (int i = 0; i < 16; ++i) bv[i] = vrow[i];  // contiguous = 2x b128
        v8f& o = (nt == 0) ? o0 : (nt == 1) ? o1 : (nt == 2) ? o2 : o3;
        o = wmma_f16(ap, bv, o);
      }
    }
    __syncthreads();
  }

  // Store merged-head bf16 output: ATT[b2][tok][h*64+d]
  for (int nt = 0; nt < 4; ++nt) {
    const v8f o = (nt == 0) ? o0 : (nt == 1) ? o1 : (nt == 2) ? o2 : o3;
    int dcol = nt * 16 + ncol;
    for (int j = 0; j < 8; ++j) {
      int tok = q0 + wid * 16 + j + hi * 8;
      ATT[((size_t)b2 * NTOK + tok) * CDIM + h * DH + dcol] = (__bf16)o[j];
    }
  }
}

// ---------------- Kernel 3: output projection (64x128 macro-tile GEMM) -----
__global__ __launch_bounds__(256) void proj_kernel(
    const __bf16* __restrict__ ATT, const float* __restrict__ w_proj,
    const float* __restrict__ b_proj, float* __restrict__ out) {
  __shared__ __bf16 As[GM * G_STR];
  __shared__ __bf16 Bs[GN * G_STR];
  const int tid  = threadIdx.x;
  const int lane = tid & 31, wid = tid >> 5;
  const int wm = wid >> 1, wn = wid & 1;
  const int m0 = blockIdx.y * GM;
  const int c0 = blockIdx.x * GN;
  const int mrow = lane & 15, hi = lane >> 4, ncol = lane & 15;
  v8f acc0 = {}, acc1 = {}, acc2 = {}, acc3 = {};
  for (int k0 = 0; k0 < CDIM; k0 += 32) {
    for (int e = tid; e < GM * 32; e += 256) {
      int r = e >> 5, c = e & 31;
      As[r * G_STR + c] = ATT[(size_t)(m0 + r) * CDIM + k0 + c];
    }
    for (int e = tid; e < GN * 32; e += 256) {
      int r = e >> 5, c = e & 31;
      Bs[r * G_STR + c] = (__bf16)w_proj[(size_t)(c0 + r) * CDIM + k0 + c];
    }
    __syncthreads();
    v16bf a;
    const __bf16* apr = &As[(wm * 16 + mrow) * G_STR];
    for (int i = 0; i < 8; ++i) {
      int k = a_kidx(i, hi);
      a[2 * i]     = apr[k];
      a[2 * i + 1] = apr[k + 1];
    }
#pragma unroll
    for (int nt = 0; nt < 4; ++nt) {
      const __bf16* bp = &Bs[(wn * 64 + nt * 16 + ncol) * G_STR + hi * 16];
      v16bf b;
      for (int i = 0; i < 16; ++i) b[i] = bp[i];
      v8f& acc = (nt == 0) ? acc0 : (nt == 1) ? acc1 : (nt == 2) ? acc2 : acc3;
      acc = wmma_bf16(a, b, acc);
    }
    __syncthreads();
  }
#pragma unroll
  for (int nt = 0; nt < 4; ++nt) {
    const v8f acc = (nt == 0) ? acc0 : (nt == 1) ? acc1 : (nt == 2) ? acc2 : acc3;
    const int col   = c0 + wn * 64 + nt * 16 + ncol;
    const float bias = b_proj[col];
    for (int j = 0; j < 8; ++j) {
      int m = m0 + wm * 16 + j + hi * 8;
      out[(size_t)m * CDIM + col] = acc[j] + bias;   // [b2][tok][c] layout
    }
  }
}

extern "C" void kernel_launch(void* const* d_in, const int* in_sizes, int n_in,
                              void* d_out, int out_size, void* d_ws, size_t ws_size,
                              hipStream_t stream) {
  const float* x_v    = (const float*)d_in[0];
  const float* x_i    = (const float*)d_in[1];
  const float* w_qkv  = (const float*)d_in[2];
  const float* b_qkv  = (const float*)d_in[3];
  const float* w_proj = (const float*)d_in[4];
  const float* b_proj = (const float*)d_in[5];
  float* out = (float*)d_out;

  const size_t per = (size_t)B2T * NH * NTOK * DH;   // 37,748,736 elems
  __bf16*   Qb  = (__bf16*)d_ws;
  __bf16*   Kb  = Qb + per;
  _Float16* Vh  = (_Float16*)(Kb + per);
  __bf16*   ATT = (__bf16*)(Vh + per);               // total ~302 MB of d_ws

  qkv_kernel<<<dim3(QKVC / GN, MTOT / GM), 256, 0, stream>>>(x_v, x_i, w_qkv,
                                                             b_qkv, Qb, Kb, Vh);
  attn_kernel<<<dim3(NTOK / 64, NH, B2T), 128, 0, stream>>>(Qb, Kb, Vh, ATT);
  proj_kernel<<<dim3(CDIM / GN, MTOT / GM), 256, 0, stream>>>(ATT, w_proj,
                                                              b_proj, out);
}
